// MultivariateTSModel_59279138619460
// MI455X (gfx1250) — matrix operand order repaired and checked by reading, not compiled, not run on hardware
//
#include <hip/hip_runtime.h>
#include <hip/hip_bf16.h>

typedef __attribute__((ext_vector_type(16))) _Float16 v16h;
typedef __attribute__((ext_vector_type(8)))  float    v8f;
typedef _Float16 f16;

#define DEVI static __device__ __forceinline__

constexpr int NB = 32, NT = 256, NC = 64, NE = 32, NH = 64;
constexpr int E1 = 205, E1P = 208;      // padded N for W1 gemm (13 tiles)
constexpr int E2 = 2048;
constexpr int KP = 224;                 // padded K (E1) for W2 gemm (7 x 32)
constexpr int CC = NC * NC;             // 4096
constexpr int ROWS = NB * NT;           // 8192

// ---------- WMMA fragment helpers (f16 16x16x32, f32 accum) ----------
// A-matrix 16x32 layout used here (contiguous k): lane L (m=L&15), halves h:
//   K = h + 16*(L>>4)
DEVI int akmap(int h, int hi) { return h + 16 * hi; }

// A operand, source stored row-major [row][k] with leading dim ld
DEVI v16h load_a(const f16* p, int row, int ld, int kb, int hi) {
  v16h f;
#pragma unroll
  for (int h = 0; h < 16; ++h) f[h] = p[row * ld + kb + akmap(h, hi)];
  return f;
}
// B operand where B = X^T (X row-major [n][k]): lanes0-15 K=0..15, lanes16-31 K=16..31
DEVI v16h load_bT(const f16* p, int n, int ld, int kb, int hi) {
  v16h f;
#pragma unroll
  for (int h = 0; h < 16; ++h) f[h] = p[n * ld + kb + 16 * hi + h];
  return f;
}
// B operand where B = X (X row-major [k][n])
DEVI v16h load_bN(const f16* p, int n, int ld, int kb, int hi) {
  v16h f;
#pragma unroll
  for (int h = 0; h < 16; ++h) f[h] = p[(kb + 16 * hi + h) * ld + n];
  return f;
}
DEVI v8f wmma32(v16h a, v16h b, v8f c) {
  return __builtin_amdgcn_wmma_f32_16x16x32_f16(false, a, false, b, (short)0, c, false, false);
}

// ---------- prep: spectral norm of Wv + f16 weight copies + zero h ----------
__global__ __launch_bounds__(256) void prep_kernel(
    const float* __restrict__ Wv, const float* __restrict__ u_sn,
    const float* __restrict__ Wih, const float* __restrict__ Whh,
    const float* __restrict__ Wq, const float* __restrict__ Wk,
    f16* wih_h, f16* whh_h, f16* wq_h, f16* wk_h, f16* wvsn_h, float* hbuf) {
  __shared__ float su[64], sv[64], st[64], sscl;
  int tid = threadIdx.x;
  for (int i = tid; i < 192 * 32; i += 256) wih_h[i] = (f16)Wih[i];
  for (int i = tid; i < 192 * 64; i += 256) whh_h[i] = (f16)Whh[i];
  for (int i = tid; i < 4096; i += 256) { wq_h[i] = (f16)Wq[i]; wk_h[i] = (f16)Wk[i]; }
  for (int i = tid; i < NB * NC * NH; i += 256) hbuf[i] = 0.f;
  if (tid < 64) su[tid] = u_sn[tid];
  __syncthreads();
  for (int it = 0; it < 5; ++it) {
    if (tid < 64) { float s = 0.f; for (int i = 0; i < 64; ++i) s += Wv[i * 64 + tid] * su[i]; st[tid] = s; }
    __syncthreads();
    if (tid == 0) { float s = 0.f; for (int i = 0; i < 64; ++i) s += st[i] * st[i]; sscl = 1.f / (sqrtf(s) + 1e-12f); }
    __syncthreads();
    if (tid < 64) sv[tid] = st[tid] * sscl;
    __syncthreads();
    if (tid < 64) { float s = 0.f; for (int j = 0; j < 64; ++j) s += Wv[tid * 64 + j] * sv[j]; st[tid] = s; }
    __syncthreads();
    if (tid == 0) { float s = 0.f; for (int i = 0; i < 64; ++i) s += st[i] * st[i]; sscl = 1.f / (sqrtf(s) + 1e-12f); }
    __syncthreads();
    if (tid < 64) su[tid] = st[tid] * sscl;
    __syncthreads();
  }
  if (tid < 64) { float s = 0.f; for (int j = 0; j < 64; ++j) s += Wv[tid * 64 + j] * sv[j]; st[tid] = s; }
  __syncthreads();
  if (tid == 0) { float s = 0.f; for (int i = 0; i < 64; ++i) s += su[i] * st[i]; sscl = 1.f / s; }
  __syncthreads();
  float inv = sscl;
  for (int i = tid; i < 4096; i += 256) wvsn_h[i] = (f16)(Wv[i] * inv);
}

// ---------- f16 conversion of W1 (pad rows to 208) and W2 (pad K to 224) ----------
__global__ __launch_bounds__(256) void convw_kernel(const float* __restrict__ W1,
                                                    const float* __restrict__ W2,
                                                    f16* W1h, f16* W2h) {
  int idx = blockIdx.x * 256 + threadIdx.x;
  const int N1 = E1P * CC;
  if (idx < N1) {
    int r = idx >> 12, c = idx & 4095;
    W1h[idx] = (f16)(r < E1 ? W1[r * CC + c] : 0.f);
  } else {
    int i2 = idx - N1;                  // < 2048*224
    int r = i2 / KP, c = i2 % KP;
    W2h[i2] = (f16)(c < E1 ? W2[r * E1 + c] : 0.f);
  }
}

// ---------- fused per-timestep kernel: GRU + QKV + attn + power iter ----------
__global__ __launch_bounds__(256) void step_kernel(
    const float* __restrict__ x, const float* __restrict__ embW, const float* __restrict__ embB,
    const float* __restrict__ b_ih, const float* __restrict__ b_hh,
    const float* __restrict__ bq, const float* __restrict__ bk, const float* __restrict__ bv,
    const float* __restrict__ pi_init,
    const f16* __restrict__ wih_h, const f16* __restrict__ whh_h,
    const f16* __restrict__ wq_h, const f16* __restrict__ wk_h, const f16* __restrict__ wvsn_h,
    float* __restrict__ hbuf, float* __restrict__ outA, int t) {
  extern __shared__ char smem[];
  f16*  s_emb = (f16*)smem;                               // 64x32   (4096 B)
  f16*  s_h   = (f16*)(smem + 4096);                      // 64x64   (8192 B)
  f16*  s_h2  = (f16*)(smem + 12288);                     // 64x64   (8192 B)
  char* G     = smem + 20480;                             // 98304 B shared region
  float* s_gi = (float*)G;                                 // 64x192
  float* s_gh = (float*)(G + 49152);                       // 64x192
  f16*  s_q   = (f16*)G;                                   // phase >=3 overlays
  f16*  s_k   = (f16*)(G + 8192);
  f16*  s_v   = (f16*)(G + 16384);
  float* s_A  = (float*)(G + 24576);                       // 64x64 f32
  f16*  s_Ah  = (f16*)(G + 40960);                         // 64x64 f16
  float* s_x  = (float*)(smem + 118784);
  float* s_y  = s_x + 64;
  float* s_scl = s_x + 128;

  const int b = blockIdx.x;
  const int tid = threadIdx.x, wv = tid >> 5, lane = tid & 31, ln = lane & 15, hi = lane >> 4;
  const float* hb = hbuf + b * (NC * NH);

  // phase 0: embed + convert h to f16
  {
    int c = tid >> 2, e0 = (tid & 3) * 8;
    float xv = x[((size_t)b * NT + t) * NC + c];
    for (int e = e0; e < e0 + 8; ++e)
      s_emb[c * NE + e] = (f16)(xv * embW[c * NE + e] + embB[c * NE + e]);
    for (int i = tid; i < NC * NH; i += 256) s_h[i] = (f16)hb[i];
  }
  __syncthreads();

  // phase 1: gh = h @ Whh^T (48 tiles) ; gi = emb @ Wih^T (48 tiles)
  for (int job = wv; job < 96; job += 8) {
    int which = job / 48, tIdx = job % 48;
    int mt = tIdx / 12, nt = tIdx % 12;
    v8f acc = {};
    if (which == 0) {
      for (int kt = 0; kt < 2; ++kt) {
        v16h a = load_a(s_h, mt * 16 + ln, 64, kt * 32, hi);
        v16h bb = load_bT(whh_h, nt * 16 + ln, 64, kt * 32, hi);
        acc = wmma32(a, bb, acc);
      }
#pragma unroll
      for (int r = 0; r < 8; ++r) s_gh[(mt * 16 + r + 8 * hi) * 192 + nt * 16 + ln] = acc[r];
    } else {
      v16h a = load_a(s_emb, mt * 16 + ln, 32, 0, hi);
      v16h bb = load_bT(wih_h, nt * 16 + ln, 32, 0, hi);
      acc = wmma32(a, bb, acc);
#pragma unroll
      for (int r = 0; r < 8; ++r) s_gi[(mt * 16 + r + 8 * hi) * 192 + nt * 16 + ln] = acc[r];
    }
  }
  __syncthreads();

  // phase 2: GRU elementwise -> h2 (f16)
  {
    int c = tid >> 2, h0 = (tid & 3) * 16;
    for (int hh = h0; hh < h0 + 16; ++hh) {
      float gr  = s_gi[c * 192 + hh]        + b_ih[hh]        + s_gh[c * 192 + hh]        + b_hh[hh];
      float gz  = s_gi[c * 192 + 64 + hh]   + b_ih[64 + hh]   + s_gh[c * 192 + 64 + hh]   + b_hh[64 + hh];
      float gin = s_gi[c * 192 + 128 + hh]  + b_ih[128 + hh];
      float ghn = s_gh[c * 192 + 128 + hh]  + b_hh[128 + hh];
      float r = 1.f / (1.f + expf(-gr));
      float z = 1.f / (1.f + expf(-gz));
      float n = tanhf(gin + r * ghn);
      float hp = hb[c * 64 + hh];
      s_h2[c * 64 + hh] = (f16)((1.f - z) * n + z * hp);
    }
  }
  __syncthreads();

  // phase 3: q/k/v = h2 @ {Wq,Wk,Wv_sn}^T  (48 tiles)
  for (int job = wv; job < 48; job += 8) {
    int which = job >> 4, sub = job & 15, mt = sub >> 2, nt = sub & 3;
    const f16* Wp = which == 0 ? wq_h : (which == 1 ? wk_h : wvsn_h);
    const float* bp = which == 0 ? bq : (which == 1 ? bk : bv);
    f16* Dp = which == 0 ? s_q : (which == 1 ? s_k : s_v);
    v8f acc = {};
    for (int kt = 0; kt < 2; ++kt) {
      v16h a = load_a(s_h2, mt * 16 + ln, 64, kt * 32, hi);
      v16h bb = load_bT(Wp, nt * 16 + ln, 64, kt * 32, hi);
      acc = wmma32(a, bb, acc);
    }
    int n = nt * 16 + ln; float bias = bp[n];
#pragma unroll
    for (int r = 0; r < 8; ++r) Dp[(mt * 16 + r + 8 * hi) * 64 + n] = (f16)(acc[r] + bias);
  }
  __syncthreads();

  // phase 4: A = tanh(q @ k^T)  (16 tiles)
  for (int job = wv; job < 16; job += 8) {
    int mt = job >> 2, nt = job & 3;
    v8f acc = {};
    for (int kt = 0; kt < 2; ++kt) {
      v16h a = load_a(s_q, mt * 16 + ln, 64, kt * 32, hi);
      v16h bb = load_bT(s_k, nt * 16 + ln, 64, kt * 32, hi);
      acc = wmma32(a, bb, acc);
    }
    int n = nt * 16 + ln;
#pragma unroll
    for (int r = 0; r < 8; ++r) s_A[(mt * 16 + r + 8 * hi) * 64 + n] = tanhf(acc[r]);
  }
  __syncthreads();

  // phase 5: power iteration (10 steps) + Rayleigh quotient
  if (tid < 64) s_x[tid] = pi_init[tid];
  __syncthreads();
  for (int it = 0; it < 10; ++it) {
    if (tid < 64) { float s = 0.f; for (int j = 0; j < 64; ++j) s += s_A[tid * 64 + j] * s_x[j]; s_y[tid] = s; }
    __syncthreads();
    if (tid == 0) { float s = 0.f; for (int j = 0; j < 64; ++j) s += s_y[j] * s_y[j]; s_scl[0] = 1.f / sqrtf(s); }
    __syncthreads();
    if (tid < 64) s_x[tid] = s_y[tid] * s_scl[0];
    __syncthreads();
  }
  if (tid < 64) { float s = 0.f; for (int j = 0; j < 64; ++j) s += s_A[tid * 64 + j] * s_x[j]; s_y[tid] = s; }
  __syncthreads();
  if (tid == 0) {
    float num = 0.f, den = 0.f;
    for (int j = 0; j < 64; ++j) { num += s_x[j] * s_y[j]; den += s_x[j] * s_x[j]; }
    s_scl[0] = 0.9f / fabsf(num / den);
  }
  __syncthreads();

  // phase 6: scale A, write A_seq (d_out), keep f16 copy
  {
    float scl = s_scl[0];
    float* outAp = outA + ((size_t)(b * NT + t)) * CC;
    for (int i = tid; i < CC; i += 256) {
      float a = s_A[i] * scl;
      outAp[i] = a;
      s_Ah[i] = (f16)a;
    }
  }
  __syncthreads();

  // phase 7: next carry = A @ v  (16 tiles)
  for (int job = wv; job < 16; job += 8) {
    int mt = job >> 2, nt = job & 3;
    v8f acc = {};
    for (int kt = 0; kt < 2; ++kt) {
      v16h a = load_a(s_Ah, mt * 16 + ln, 64, kt * 32, hi);
      v16h bb = load_bN(s_v, nt * 16 + ln, 64, kt * 32, hi);
      acc = wmma32(a, bb, acc);
    }
    int n = nt * 16 + ln;
    float* ho = hbuf + b * (NC * NH);
#pragma unroll
    for (int r = 0; r < 8; ++r) ho[(mt * 16 + r + 8 * hi) * 64 + n] = acc[r];
  }
}

// ---------- mean over T ----------
__global__ __launch_bounds__(256) void mean_kernel(const float* __restrict__ outA, float* m_ws) {
  int b = blockIdx.x, cc = blockIdx.y * 256 + threadIdx.x;
  const float* p = outA + (size_t)b * NT * CC + cc;
  float s = 0.f;
  for (int t = 0; t < NT; ++t) s += p[(size_t)t * CC];
  m_ws[b * CC + cc] = s * (1.f / NT);
}

// ---------- e = (flat * mean) @ W1^T + b1 : WMMA, K=4096 ----------
__global__ __launch_bounds__(256) void egemm_kernel(const float* __restrict__ outA,
                                                    const float* __restrict__ m_ws,
                                                    const f16* __restrict__ W1h,
                                                    const float* __restrict__ b1,
                                                    float* __restrict__ e_ws) {
  int row0 = blockIdx.x * 64;
  int tid = threadIdx.x, wv = tid >> 5, lane = tid & 31, ln = lane & 15, hi = lane >> 4;
  int nt0 = wv, nt1 = wv + 8;
  bool has1 = nt1 < 13;
  for (int mt = 0; mt < 4; ++mt) {
    int row = row0 + mt * 16 + ln;
    const float* Ap = outA + (size_t)row * CC;
    const float* Mp = m_ws + (size_t)(row >> 8) * CC;
    v8f a0 = {}, a1 = {};
    for (int kt = 0; kt < 128; ++kt) {
      int kb = kt * 32;
      v16h af;
#pragma unroll
      for (int h = 0; h < 16; ++h) { int k = kb + akmap(h, hi); af[h] = (f16)(Ap[k] * Mp[k]); }
      v16h bf0 = load_bT(W1h, nt0 * 16 + ln, CC, kb, hi);
      a0 = wmma32(af, bf0, a0);
      if (has1) { v16h bf1 = load_bT(W1h, nt1 * 16 + ln, CC, kb, hi); a1 = wmma32(af, bf1, a1); }
    }
#pragma unroll
    for (int r = 0; r < 8; ++r) {
      int m = mt * 16 + r + 8 * hi;
      int n = nt0 * 16 + ln;
      e_ws[(size_t)(row0 + m) * E1P + n] = a0[r] + (n < E1 ? b1[n] : 0.f);
      if (has1) { int n1 = nt1 * 16 + ln; e_ws[(size_t)(row0 + m) * E1P + n1] = a1[r] + (n1 < E1 ? b1[n1] : 0.f); }
    }
  }
}

// ---------- BatchNorm stats (biased var over 8192 rows) ----------
__global__ __launch_bounds__(256) void bnstats_kernel(const float* __restrict__ e_ws,
                                                      const float* __restrict__ gamma,
                                                      const float* __restrict__ beta,
                                                      float* scale_ws, float* shift_ws) {
  int j = blockIdx.x;   // < 205
  __shared__ float s1[256], s2[256];
  int tid = threadIdx.x;
  float a = 0.f, b = 0.f;
  for (int r = tid; r < ROWS; r += 256) { float v = e_ws[(size_t)r * E1P + j]; a += v; b += v * v; }
  s1[tid] = a; s2[tid] = b; __syncthreads();
  for (int s = 128; s > 0; s >>= 1) {
    if (tid < s) { s1[tid] += s1[tid + s]; s2[tid] += s2[tid + s]; }
    __syncthreads();
  }
  if (tid == 0) {
    float mu = s1[0] / (float)ROWS, var = s2[0] / (float)ROWS - mu * mu;
    float sc = gamma[j] * rsqrtf(var + 1e-5f);
    scale_ws[j] = sc; shift_ws[j] = beta[j] - mu * sc;
  }
}

// ---------- BN + ReLU -> f16, K padded to 224 ----------
__global__ __launch_bounds__(256) void bnrelu_kernel(const float* __restrict__ e_ws,
                                                     const float* __restrict__ scale_ws,
                                                     const float* __restrict__ shift_ws,
                                                     f16* __restrict__ eact) {
  int idx = blockIdx.x * 256 + threadIdx.x;     // < 8192*224
  int row = idx / KP, col = idx % KP;
  float v = 0.f;
  if (col < E1) v = fmaxf(e_ws[(size_t)row * E1P + col] * scale_ws[col] + shift_ws[col], 0.f);
  eact[idx] = (f16)v;
}

// ---------- fused: relu(eact @ W2^T + b2) . W3  -> a_accum (deterministic) ----------
__global__ __launch_bounds__(256) void hiddot_kernel(const f16* __restrict__ eact,
                                                     const f16* __restrict__ W2h,
                                                     const float* __restrict__ b2,
                                                     const float* __restrict__ w3,
                                                     float* __restrict__ a_accum) {
  __shared__ float s_part[8 * 64];
  int row0 = blockIdx.x * 64;
  int tid = threadIdx.x, wv = tid >> 5, lane = tid & 31, ln = lane & 15, hi = lane >> 4;
  for (int mt = 0; mt < 4; ++mt) {
    float p[8];
#pragma unroll
    for (int r = 0; r < 8; ++r) p[r] = 0.f;
    for (int nt = wv; nt < 128; nt += 8) {
      v8f acc = {};
      for (int kt = 0; kt < 7; ++kt) {
        v16h a = load_a(eact, row0 + mt * 16 + ln, KP, kt * 32, hi);
        v16h bb = load_bT(W2h, nt * 16 + ln, KP, kt * 32, hi);
        acc = wmma32(a, bb, acc);
      }
      int n = nt * 16 + ln;
      float bias = b2[n], w3v = w3[n];
#pragma unroll
      for (int r = 0; r < 8; ++r) p[r] += fmaxf(acc[r] + bias, 0.f) * w3v;
    }
#pragma unroll
    for (int r = 0; r < 8; ++r) {
      float v = p[r];
      for (int m = 1; m < 16; m <<= 1) v += __shfl_xor(v, m, 32);
      if (ln == 0) s_part[wv * 64 + mt * 16 + r + 8 * hi] = v;
    }
  }
  __syncthreads();
  if (tid < 64) {
    float s = 0.f;
    for (int w = 0; w < 8; ++w) s += s_part[w * 64 + tid];
    a_accum[row0 + tid] = s;   // b3 omitted: softmax shift-invariant
  }
}

// ---------- softmax over T per batch ----------
__global__ __launch_bounds__(256) void softmax_kernel(const float* __restrict__ a_accum,
                                                      float* __restrict__ a_soft) {
  int b = blockIdx.x, t = threadIdx.x;
  __shared__ float sh[256];
  float v = a_accum[b * NT + t];
  sh[t] = v; __syncthreads();
  for (int s = 128; s > 0; s >>= 1) { if (t < s) sh[t] = fmaxf(sh[t], sh[t + s]); __syncthreads(); }
  float mx = sh[0];
  __syncthreads();
  float e = expf(v - mx);
  sh[t] = e; __syncthreads();
  for (int s = 128; s > 0; s >>= 1) { if (t < s) sh[t] += sh[t + s]; __syncthreads(); }
  a_soft[b * NT + t] = e / sh[0];
}

// ---------- DNC = sum_t flat * a ----------
__global__ __launch_bounds__(256) void dnc_kernel(const float* __restrict__ outA,
                                                  const float* __restrict__ a_soft,
                                                  float* __restrict__ out_dnc) {
  int b = blockIdx.x, cc = blockIdx.y * 256 + threadIdx.x;
  __shared__ float sa[256];
  sa[threadIdx.x] = a_soft[b * NT + threadIdx.x];
  __syncthreads();
  const float* p = outA + (size_t)b * NT * CC + cc;
  float s = 0.f;
  for (int t = 0; t < NT; ++t) s += p[(size_t)t * CC] * sa[t];
  out_dnc[b * CC + cc] = s;
}

// ---------- classifier ----------
__global__ void logits_kernel(const float* __restrict__ dnc, const float* __restrict__ Wc,
                              const float* __restrict__ bc, float* __restrict__ out_logits) {
  int i = threadIdx.x;
  if (i >= 64) return;
  int b = i >> 1, o = i & 1;
  const float* d = dnc + b * CC;
  const float* w = Wc + o * CC;
  float s = 0.f;
  for (int k = 0; k < CC; ++k) s += d[k] * w[k];
  out_logits[i] = s + bc[o];
}

// =====================================================================
extern "C" void kernel_launch(void* const* d_in, const int* in_sizes, int n_in,
                              void* d_out, int out_size, void* d_ws, size_t ws_size,
                              hipStream_t stream) {
  (void)in_sizes; (void)n_in; (void)out_size; (void)ws_size;
  const float* x      = (const float*)d_in[0];
  const float* embW   = (const float*)d_in[1];
  const float* embB   = (const float*)d_in[2];
  const float* W_ih   = (const float*)d_in[3];
  const float* W_hh   = (const float*)d_in[4];
  const float* b_ih   = (const float*)d_in[5];
  const float* b_hh   = (const float*)d_in[6];
  const float* Wq     = (const float*)d_in[7];
  const float* bq     = (const float*)d_in[8];
  const float* Wk     = (const float*)d_in[9];
  const float* bk     = (const float*)d_in[10];
  const float* Wv     = (const float*)d_in[11];
  const float* bv     = (const float*)d_in[12];
  const float* u_sn   = (const float*)d_in[13];
  const float* pi_init= (const float*)d_in[14];
  const float* W1     = (const float*)d_in[15];
  const float* b1     = (const float*)d_in[16];
  const float* gamma  = (const float*)d_in[17];
  const float* beta   = (const float*)d_in[18];
  const float* W2     = (const float*)d_in[19];
  const float* b2     = (const float*)d_in[20];
  const float* W3     = (const float*)d_in[21];
  const float* b3     = (const float*)d_in[22]; (void)b3;  // cancels in softmax
  const float* Wc     = (const float*)d_in[23];
  const float* bc     = (const float*)d_in[24];

  float* out        = (float*)d_out;
  float* out_logits = out;                       // 64
  float* out_dnc    = out + 64;                  // 131072
  float* out_A      = out + 64 + NB * CC;        // 32*256*4096

  char* ws = (char*)d_ws;
  size_t off = 0;
  auto carve = [&](size_t bytes) { void* p = ws + off; off = (off + bytes + 255) & ~(size_t)255; return p; };
  float* hbuf   = (float*)carve((size_t)NB * NC * NH * 4);
  f16* wih_h    = (f16*)carve(192 * 32 * 2);
  f16* whh_h    = (f16*)carve(192 * 64 * 2);
  f16* wq_h     = (f16*)carve(4096 * 2);
  f16* wk_h     = (f16*)carve(4096 * 2);
  f16* wvsn_h   = (f16*)carve(4096 * 2);
  f16* W1h      = (f16*)carve((size_t)E1P * CC * 2);
  f16* W2h      = (f16*)carve((size_t)E2 * KP * 2);
  float* m_ws   = (float*)carve((size_t)NB * CC * 4);
  float* e_ws   = (float*)carve((size_t)ROWS * E1P * 4);
  float* scale_ws = (float*)carve(1024);
  float* shift_ws = (float*)carve(1024);
  f16* eact     = (f16*)carve((size_t)ROWS * KP * 2);
  float* a_accum= (float*)carve((size_t)ROWS * 4);
  float* a_soft = (float*)carve((size_t)ROWS * 4);

  prep_kernel<<<1, 256, 0, stream>>>(Wv, u_sn, W_ih, W_hh, Wq, Wk,
                                     wih_h, whh_h, wq_h, wk_h, wvsn_h, hbuf);
  convw_kernel<<<(E1P * CC + E2 * KP) / 256, 256, 0, stream>>>(W1, W2, W1h, W2h);

  const size_t stepShm = 4096 + 8192 + 8192 + 98304 + 1024;   // ~117 KB dynamic LDS
  for (int t = 0; t < NT; ++t)
    step_kernel<<<NB, 256, stepShm, stream>>>(x, embW, embB, b_ih, b_hh, bq, bk, bv,
                                              pi_init, wih_h, whh_h, wq_h, wk_h, wvsn_h,
                                              hbuf, out_A, t);

  mean_kernel<<<dim3(NB, 16), 256, 0, stream>>>(out_A, m_ws);
  egemm_kernel<<<ROWS / 64, 256, 0, stream>>>(out_A, m_ws, W1h, b1, e_ws);
  bnstats_kernel<<<E1, 256, 0, stream>>>(e_ws, gamma, beta, scale_ws, shift_ws);
  bnrelu_kernel<<<ROWS * KP / 256, 256, 0, stream>>>(e_ws, scale_ws, shift_ws, eact);
  hiddot_kernel<<<ROWS / 64, 256, 0, stream>>>(eact, W2h, b2, W3, a_accum);
  softmax_kernel<<<NB, 256, 0, stream>>>(a_accum, a_soft);
  dnc_kernel<<<dim3(NB, 16), 256, 0, stream>>>(out_A, a_soft, out_dnc);
  logits_kernel<<<1, 64, 0, stream>>>(out_dnc, Wc, bc, out_logits);
}
